// TopFrag_90142773609217
// MI455X (gfx1250) — matrix operand order repaired and checked by reading, not compiled, not run on hardware
//
#include <hip/hip_runtime.h>
#include <math.h>

// ---------------- model constants ----------------
#define DIM      1024
#define NHEADS   16
#define HEADD    64
#define FFD      4096
#define MOE_FF   1024
#define NEXP     8
#define SHARED_FF 2048
#define NLAYERS  3
#define NDENSE   1
#define SMI_V    512
#define BB       8
#define S_SRC    512
#define S_TGT    128
#define NE       (BB * S_SRC)   // 4096 encoder tokens
#define NDT      (BB * S_TGT)   // 1024 decoder tokens
#define EPS_RMS  1.1920929e-07f
#define NEG_INF9 (-1e9f)

// ---------------- WMMA types ----------------
typedef __attribute__((ext_vector_type(16))) __bf16 v16bf;
typedef __attribute__((ext_vector_type(8)))  float  v8f;
struct U32x8 { uint4 lo, hi; };

__device__ inline v16bf load_frag(const __bf16* p, int gap_halves) {
  U32x8 u;
  u.lo = *(const uint4*)(p);
  u.hi = *(const uint4*)(p + gap_halves);
  return __builtin_bit_cast(v16bf, u);
}

// =====================================================================
// GEMM: C[N,M] = A[N,K] @ W[M,K]^T + bias[M]
// bf16 WMMA 16x16x32, fp32 accumulate. N%128==0, M%128==0, K%32==0.
// 128x128 block tile, 8 wave32s, each wave owns 32x64 (2x4 WMMA tiles).
// Ping-pong LDS double buffer: global loads of chunk k+1 overlap WMMAs
// of chunk k; one barrier per K-chunk.
// =====================================================================
#define BM 128
#define BN 128
#define BK 32
#define LDH 40   // halves per staged row (pad vs 32 to spread banks)

__global__ __launch_bounds__(256) void k_gemm(const float* __restrict__ A,
                                              const float* __restrict__ W,
                                              const float* __restrict__ bias,
                                              float* __restrict__ C,
                                              int N, int M, int K) {
  __shared__ __align__(16) __bf16 sA[2][BM * LDH];
  __shared__ __align__(16) __bf16 sB[2][BN * LDH];

  const int t    = threadIdx.x;
  const int lane = t & 31;
  const int wave = t >> 5;
  const int wm   = (wave & 3) * 32;   // 0..96
  const int wn   = (wave >> 2) * 64;  // 0 or 64
  const int n0   = blockIdx.y * BM;
  const int m0   = blockIdx.x * BN;

  v8f acc[2][4] = {};

  const int ar = lane & 15;       // row-in-tile (A) / col-in-tile (B)
  const int ah = lane >> 4;       // half selector

  // per-thread staging: 4 float4 of A + 4 float4 of B per K-chunk
  float4 regA[4], regB[4];
  const int sr = t >> 3;            // staged row 0..31 step, idx>>3 below
  (void)sr;

  auto gload = [&](int k0) {
#pragma unroll
    for (int i = 0; i < 4; ++i) {
      int idx = t + i * 256;          // float4 index, 1024 per matrix
      int r = idx >> 3, c = (idx & 7) << 2;
      regA[i] = *(const float4*)(A + (size_t)(n0 + r) * K + k0 + c);
      regB[i] = *(const float4*)(W + (size_t)(m0 + r) * K + k0 + c);
    }
  };
  auto sstore = [&](int p) {
#pragma unroll
    for (int i = 0; i < 4; ++i) {
      int idx = t + i * 256;
      int r = idx >> 3, c = (idx & 7) << 2;
      __bf16* da = &sA[p][r * LDH + c];
      da[0] = (__bf16)regA[i].x; da[1] = (__bf16)regA[i].y;
      da[2] = (__bf16)regA[i].z; da[3] = (__bf16)regA[i].w;
      __bf16* db = &sB[p][r * LDH + c];
      db[0] = (__bf16)regB[i].x; db[1] = (__bf16)regB[i].y;
      db[2] = (__bf16)regB[i].z; db[3] = (__bf16)regB[i].w;
    }
  };

  gload(0);
  sstore(0);
  __syncthreads();

  const int nk = K / BK;
  int p = 0;
  for (int kc = 0; kc < nk; ++kc) {
    if (kc + 1 < nk) gload((kc + 1) * BK);   // prefetch next chunk to regs

    v16bf afrag[2], bfrag[4];
#pragma unroll
    for (int i = 0; i < 2; ++i)
      afrag[i] = load_frag(&sA[p][(wm + i * 16 + ar) * LDH + ah * 8], 16);
#pragma unroll
    for (int j = 0; j < 4; ++j)
      bfrag[j] = load_frag(&sB[p][(wn + j * 16 + ar) * LDH + ah * 8], 16);

#pragma unroll
    for (int i = 0; i < 2; ++i)
#pragma unroll
      for (int j = 0; j < 4; ++j)
        acc[i][j] = __builtin_amdgcn_wmma_f32_16x16x32_bf16(
            false, afrag[i], false, bfrag[j], (short)0, acc[i][j], false, false);

    if (kc + 1 < nk) sstore(p ^ 1);          // fill the other buffer
    __syncthreads();
    p ^= 1;
  }

  // ---- epilogue: C/D layout: row = v + 8*(lane>=16), col = lane&15 ----
  const int cm = (lane >> 4) * 8;
  const int cn = lane & 15;
#pragma unroll
  for (int i = 0; i < 2; ++i)
#pragma unroll
    for (int j = 0; j < 4; ++j) {
      int col = m0 + wn + j * 16 + cn;
      float bv = bias ? bias[col] : 0.0f;
#pragma unroll
      for (int v = 0; v < 8; ++v) {
        int row = n0 + wm + i * 16 + cm + v;
        C[(size_t)row * M + col] = acc[i][j][v] + bv;
      }
    }
}

// =====================================================================
// Embedding + sinusoidal positional encoding
// =====================================================================
__global__ void k_embed(float* __restrict__ out, const float* __restrict__ emb,
                        const int* __restrict__ toks, int Ntok, int L) {
  int idx = blockIdx.x * blockDim.x + threadIdx.x;
  if (idx >= Ntok * DIM) return;
  int n = idx >> 10, dd = idx & 1023;
  int pos = n % L;
  int tok = toks[n];
  float freq = expf(-(float)(dd & ~1) * (logf(10000.0f) / (float)DIM));
  float ang  = (float)pos * freq;
  float pe   = (dd & 1) ? cosf(ang) : sinf(ang);
  out[idx] = emb[(size_t)tok * DIM + dd] + pe;
}

// =====================================================================
// RMSNorm (row = block, 256 threads, DIM=1024)
// =====================================================================
__global__ __launch_bounds__(256) void k_rmsnorm(float* __restrict__ out,
                                                 const float* __restrict__ x,
                                                 const float* __restrict__ w) {
  int row = blockIdx.x, t = threadIdx.x;
  float4 v = ((const float4*)(x + (size_t)row * DIM))[t];
  float ss = v.x * v.x + v.y * v.y + v.z * v.z + v.w * v.w;
#pragma unroll
  for (int o = 16; o > 0; o >>= 1) ss += __shfl_xor(ss, o, 32);
  __shared__ float red[8];
  if ((t & 31) == 0) red[t >> 5] = ss;
  __syncthreads();
  float tot = red[0] + red[1] + red[2] + red[3] + red[4] + red[5] + red[6] + red[7];
  float sc = rsqrtf(tot * (1.0f / (float)DIM) + EPS_RMS);
  float4 wv = ((const float4*)w)[t];
  float4 o;
  o.x = v.x * sc * wv.x; o.y = v.y * sc * wv.y;
  o.z = v.z * sc * wv.z; o.w = v.w * sc * wv.w;
  ((float4*)(out + (size_t)row * DIM))[t] = o;
}

// =====================================================================
// Attention core: online softmax over keys; one thread = one query row/head
// grid = (Lq/64, NHEADS, B), block = 64
// =====================================================================
__global__ __launch_bounds__(64) void k_attn(const float* __restrict__ Q,
                                             const float* __restrict__ K,
                                             const float* __restrict__ V,
                                             float* __restrict__ O,
                                             const int* __restrict__ toks,
                                             int Lq, int Lk, int causal) {
  int qi = blockIdx.x * 64 + threadIdx.x;   // position within target sequence
  int h = blockIdx.y, b = blockIdx.z;
  const float4* qr = (const float4*)(Q + (size_t)(b * Lq + qi) * DIM + h * HEADD);
  float4 q4[16], a4[16];
#pragma unroll
  for (int i = 0; i < 16; ++i) { q4[i] = qr[i]; a4[i] = make_float4(0, 0, 0, 0); }
  float m = -INFINITY, l = 0.0f;
  for (int kk = 0; kk < Lk; ++kk) {
    const float4* kr = (const float4*)(K + (size_t)(b * Lk + kk) * DIM + h * HEADD);
    float s = 0.0f;
#pragma unroll
    for (int i = 0; i < 16; ++i) {
      float4 kv = kr[i];
      s += q4[i].x * kv.x + q4[i].y * kv.y + q4[i].z * kv.z + q4[i].w * kv.w;
    }
    s *= 0.125f;                       // 1/sqrt(64)
    if (toks[b * Lk + kk] == 0) s = NEG_INF9;
    if (causal && kk > qi) s = NEG_INF9;
    float mn = fmaxf(m, s);
    float corr = expf(m - mn);
    float pr = expf(s - mn);
    l = l * corr + pr;
    const float4* vr = (const float4*)(V + (size_t)(b * Lk + kk) * DIM + h * HEADD);
#pragma unroll
    for (int i = 0; i < 16; ++i) {
      float4 vv = vr[i];
      a4[i].x = a4[i].x * corr + pr * vv.x;
      a4[i].y = a4[i].y * corr + pr * vv.y;
      a4[i].z = a4[i].z * corr + pr * vv.z;
      a4[i].w = a4[i].w * corr + pr * vv.w;
    }
    m = mn;
  }
  float inv = 1.0f / l;
  float4* orow = (float4*)(O + (size_t)(b * Lq + qi) * DIM + h * HEADD);
#pragma unroll
  for (int i = 0; i < 16; ++i) {
    float4 o;
    o.x = a4[i].x * inv; o.y = a4[i].y * inv;
    o.z = a4[i].z * inv; o.w = a4[i].w * inv;
    orow[i] = o;
  }
}

// =====================================================================
// Elementwise helpers
// =====================================================================
__global__ void k_add(float* __restrict__ c, const float* __restrict__ a,
                      const float* __restrict__ b, int n) {
  int i = blockIdx.x * blockDim.x + threadIdx.x;
  if (i < n) c[i] = a[i] + b[i];
}

__global__ void k_mul_silu(float* __restrict__ a, const float* __restrict__ b, int n) {
  int i = blockIdx.x * blockDim.x + threadIdx.x;
  if (i >= n) return;
  float h = a[i] * b[i];
  a[i] = h / (1.0f + expf(-h));
}

__global__ void k_axpy_rows(float* __restrict__ y, const float* __restrict__ x,
                            const float* __restrict__ g, int e, int n) {
  int i = blockIdx.x * blockDim.x + threadIdx.x;
  if (i >= n) return;
  int row = i >> 10;
  y[i] += g[row * NEXP + e] * x[i];
}

// MoE gate logits: [N, 8] = x @ gate_w^T  (M=8 too small for WMMA tiling)
__global__ void k_gate_logits(const float* __restrict__ x, const float* __restrict__ gw,
                              float* __restrict__ logits, int N) {
  int idx = blockIdx.x * blockDim.x + threadIdx.x;
  if (idx >= N * NEXP) return;
  int n = idx >> 3, e = idx & 7;
  const float4* xr = (const float4*)(x + (size_t)n * DIM);
  const float4* wr = (const float4*)(gw + (size_t)e * DIM);
  float s = 0.0f;
  for (int i = 0; i < DIM / 4; ++i) {
    float4 a = xr[i], b = wr[i];
    s += a.x * b.x + a.y * b.y + a.z * b.z + a.w * b.w;
  }
  logits[idx] = s;
}

// top-2 selection (first-occurrence tie break), sigmoid weights, normalize
__global__ void k_moe_gate(const float* __restrict__ logits, const float* __restrict__ gb,
                           float* __restrict__ dense, int N) {
  int n = blockIdx.x * blockDim.x + threadIdx.x;
  if (n >= N) return;
  float lg[NEXP], sel[NEXP];
#pragma unroll
  for (int e = 0; e < NEXP; ++e) { lg[e] = logits[n * NEXP + e]; sel[e] = lg[e] + gb[e]; }
  int i0 = 0; float m0 = sel[0];
#pragma unroll
  for (int e = 1; e < NEXP; ++e) if (sel[e] > m0) { m0 = sel[e]; i0 = e; }
  int i1 = -1; float m1 = -INFINITY;
#pragma unroll
  for (int e = 0; e < NEXP; ++e) if (e != i0 && sel[e] > m1) { m1 = sel[e]; i1 = e; }
  float w0 = 1.0f / (1.0f + expf(-lg[i0]));
  float w1 = 1.0f / (1.0f + expf(-lg[i1]));
  float inv = 1.0f / (w0 + w1);
#pragma unroll
  for (int e = 0; e < NEXP; ++e)
    dense[n * NEXP + e] = (e == i0) ? w0 * inv : (e == i1) ? w1 * inv : 0.0f;
}

// log-softmax over rows of SMI_V=512, in place. block=128
__global__ __launch_bounds__(128) void k_logsoftmax(float* __restrict__ X) {
  int row = blockIdx.x, t = threadIdx.x;
  float4* r = (float4*)(X + (size_t)row * SMI_V);
  float4 v = r[t];
  float mx = fmaxf(fmaxf(v.x, v.y), fmaxf(v.z, v.w));
#pragma unroll
  for (int o = 16; o > 0; o >>= 1) mx = fmaxf(mx, __shfl_xor(mx, o, 32));
  __shared__ float redm[4], reds[4];
  if ((t & 31) == 0) redm[t >> 5] = mx;
  __syncthreads();
  mx = fmaxf(fmaxf(redm[0], redm[1]), fmaxf(redm[2], redm[3]));
  float s = expf(v.x - mx) + expf(v.y - mx) + expf(v.z - mx) + expf(v.w - mx);
#pragma unroll
  for (int o = 16; o > 0; o >>= 1) s += __shfl_xor(s, o, 32);
  if ((t & 31) == 0) reds[t >> 5] = s;
  __syncthreads();
  s = reds[0] + reds[1] + reds[2] + reds[3];
  float lg = mx + logf(s);
  v.x -= lg; v.y -= lg; v.z -= lg; v.w -= lg;
  r[t] = v;
}

// =====================================================================
// Host-side orchestration
// =====================================================================
namespace {

struct Cursor {
  void* const* din;
  int cur;
  const float* f() { return (const float*)din[cur++]; }
};

struct LinP  { const float *b, *w; };
struct AttnP { const float *bk, *bo, *bq, *bv, *wk, *wo, *wq, *wv; };
struct FFdP  { LinP l1, l2, l3; };
struct MoeP  { const float *b1, *b2, *b3, *gate_b, *gate_w; FFdP shared;
               const float *w1, *w2, *w3; };
struct FFAny { bool is_moe; FFdP d; MoeP m; };
struct EncP  { AttnP attn; FFAny ff; const float *n1, *n2; };
struct DecP  { AttnP cattn; FFAny ff; const float *n1, *n2, *n3; AttnP sattn; };

LinP  getLin(Cursor& p)  { LinP l; l.b = p.f(); l.w = p.f(); return l; }
AttnP getAttn(Cursor& p) { AttnP a; a.bk=p.f(); a.bo=p.f(); a.bq=p.f(); a.bv=p.f();
                           a.wk=p.f(); a.wo=p.f(); a.wq=p.f(); a.wv=p.f(); return a; }
FFdP  getFFd(Cursor& p)  { FFdP f; f.l1=getLin(p); f.l2=getLin(p); f.l3=getLin(p); return f; }
MoeP  getMoe(Cursor& p)  { MoeP m; m.b1=p.f(); m.b2=p.f(); m.b3=p.f();
                           m.gate_b=p.f(); m.gate_w=p.f(); m.shared=getFFd(p);
                           m.w1=p.f(); m.w2=p.f(); m.w3=p.f(); return m; }
FFAny getFF(Cursor& p, bool moe) { FFAny f; f.is_moe = moe;
                                   if (moe) f.m = getMoe(p); else f.d = getFFd(p); return f; }
DecP  getDec(Cursor& p, bool moe) { DecP d; d.cattn=getAttn(p); d.ff=getFF(p,moe);
                                    d.n1=p.f(); d.n2=p.f(); d.n3=p.f(); d.sattn=getAttn(p); return d; }
EncP  getEnc(Cursor& p, bool moe) { EncP e; e.attn=getAttn(p); e.ff=getFF(p,moe);
                                    e.n1=p.f(); e.n2=p.f(); return e; }

inline void gemm(hipStream_t s, const float* A, const float* W, const float* bias,
                 float* C, int N, int M, int K) {
  dim3 g(M / BN, N / BM);
  k_gemm<<<g, 256, 0, s>>>(A, W, bias, C, N, M, K);
}
inline void ew(int n, dim3& g) { g = dim3((n + 255) / 256); }

struct WS {
  float *e, *d, *x1, *x2, *h, *q, *k, *v, *att, *tmp, *y, *ff1, *ff3, *logits, *gw;
};

void run_mha(hipStream_t s, const AttnP& a, const float* xq, int Nq, int Lq,
             const float* xkv, int Nk, int Lk, const int* toks, int causal,
             float* out, const WS& w) {
  gemm(s, xq,  a.wq, a.bq, w.q, Nq, DIM, DIM);
  gemm(s, xkv, a.wk, a.bk, w.k, Nk, DIM, DIM);
  gemm(s, xkv, a.wv, a.bv, w.v, Nk, DIM, DIM);
  dim3 g(Lq / 64, NHEADS, BB);
  k_attn<<<g, 64, 0, s>>>(w.q, w.k, w.v, w.att, toks, Lq, Lk, causal);
  gemm(s, w.att, a.wo, a.bo, out, Nq, DIM, DIM);
}

void run_ffn(hipStream_t s, const FFdP& f, const float* h, int N, int F,
             float* out, const WS& w) {
  gemm(s, h, f.l1.w, f.l1.b, w.ff1, N, F, DIM);
  gemm(s, h, f.l3.w, f.l3.b, w.ff3, N, F, DIM);
  dim3 g; ew(N * F, g);
  k_mul_silu<<<g, 256, 0, s>>>(w.ff1, w.ff3, N * F);
  gemm(s, w.ff1, f.l2.w, f.l2.b, out, N, DIM, F);
}

void run_moe(hipStream_t s, const MoeP& m, const float* h, int N,
             float* out, const WS& w) {
  dim3 g;
  ew(N * NEXP, g);
  k_gate_logits<<<g, 256, 0, s>>>(h, m.gate_w, w.logits, N);
  ew(N, g);
  k_moe_gate<<<g, 256, 0, s>>>(w.logits, m.gate_b, w.gw, N);
  // shared FFN writes `out` first (no separate zero pass)
  run_ffn(s, m.shared, h, N, SHARED_FF, out, w);
  for (int e = 0; e < NEXP; ++e) {
    gemm(s, h, m.w1 + (size_t)e * MOE_FF * DIM, m.b1 + (size_t)e * MOE_FF,
         w.ff1, N, MOE_FF, DIM);
    gemm(s, h, m.w3 + (size_t)e * MOE_FF * DIM, m.b3 + (size_t)e * MOE_FF,
         w.ff3, N, MOE_FF, DIM);
    ew(N * MOE_FF, g);
    k_mul_silu<<<g, 256, 0, s>>>(w.ff1, w.ff3, N * MOE_FF);
    gemm(s, w.ff1, m.w2 + (size_t)e * DIM * MOE_FF, m.b2 + (size_t)e * DIM,
         w.tmp, N, DIM, MOE_FF);
    ew(N * DIM, g);
    k_axpy_rows<<<g, 256, 0, s>>>(out, w.tmp, w.gw, e, N * DIM);
  }
}

} // namespace

extern "C" void kernel_launch(void* const* d_in, const int* in_sizes, int n_in,
                              void* d_out, int out_size, void* d_ws, size_t ws_size,
                              hipStream_t stream) {
  // ---- locate src / tgt / params (handles insertion-order or sorted-order) ----
  const int* src;
  const int* tgt;
  int pbase;
  if (in_sizes[0] == NE && in_sizes[1] == NDT) {   // src, tgt, params...
    src = (const int*)d_in[0];
    tgt = (const int*)d_in[1];
    pbase = 2;
  } else {                                          // params..., src, tgt
    pbase = 0;
    src = (const int*)d_in[n_in - 2];
    tgt = (const int*)d_in[n_in - 1];
  }

  Cursor cur{d_in, pbase};
  // params sorted-key order: d_norm, layers[0..2]{dec,enc}, out{b,w}, pro_emb, smi_emb
  const float* d_norm = cur.f();
  DecP dec[NLAYERS]; EncP enc[NLAYERS];
  for (int l = 0; l < NLAYERS; ++l) {
    bool moe = (l >= NDENSE);
    dec[l] = getDec(cur, moe);
    enc[l] = getEnc(cur, moe);
  }
  LinP outp = getLin(cur);
  const float* pro_emb = cur.f();
  const float* smi_emb = cur.f();

  // ---- workspace carve-up (floats) ----
  WS w;
  {
    float* base = (float*)d_ws;
    size_t o = 0;
    auto alloc = [&](size_t n) { float* p = base + o; o += n; return p; };
    w.e    = alloc((size_t)NE * DIM);
    w.d    = alloc((size_t)NDT * DIM);
    w.x1   = alloc((size_t)NE * DIM);
    w.x2   = alloc((size_t)NE * DIM);
    w.h    = alloc((size_t)NE * DIM);
    w.q    = alloc((size_t)NE * DIM);
    w.k    = alloc((size_t)NE * DIM);
    w.v    = alloc((size_t)NE * DIM);
    w.att  = alloc((size_t)NE * DIM);
    w.tmp  = alloc((size_t)NE * DIM);
    w.y    = alloc((size_t)NE * DIM);
    w.ff1  = alloc((size_t)NE * FFD);
    w.ff3  = alloc((size_t)NE * FFD);
    w.logits = alloc((size_t)NE * NEXP);
    w.gw     = alloc((size_t)NE * NEXP);
  }

  dim3 g;
  // ---- embeddings + PE ----
  ew(NE * DIM, g);
  k_embed<<<g, 256, 0, stream>>>(w.e, pro_emb, src, NE, S_SRC);
  ew(NDT * DIM, g);
  k_embed<<<g, 256, 0, stream>>>(w.d, smi_emb, tgt, NDT, S_TGT);

  for (int l = 0; l < NLAYERS; ++l) {
    // ================= encoder block =================
    k_rmsnorm<<<NE, 256, 0, stream>>>(w.h, w.e, enc[l].n1);
    run_mha(stream, enc[l].attn, w.h, NE, S_SRC, w.h, NE, S_SRC, src, 0, w.tmp, w);
    ew(NE * DIM, g);
    k_add<<<g, 256, 0, stream>>>(w.x1, w.e, w.tmp, NE * DIM);
    k_rmsnorm<<<NE, 256, 0, stream>>>(w.h, w.x1, enc[l].n2);
    if (enc[l].ff.is_moe) run_moe(stream, enc[l].ff.m, w.h, NE, w.y, w);
    else                  run_ffn(stream, enc[l].ff.d, w.h, NE, FFD, w.y, w);
    ew(NE * DIM, g);
    k_add<<<g, 256, 0, stream>>>(w.e, w.x1, w.y, NE * DIM);

    // ================= decoder block =================
    k_rmsnorm<<<NDT, 256, 0, stream>>>(w.h, w.d, dec[l].n1);
    run_mha(stream, dec[l].sattn, w.h, NDT, S_TGT, w.h, NDT, S_TGT, tgt, 1, w.tmp, w);
    ew(NDT * DIM, g);
    k_add<<<g, 256, 0, stream>>>(w.x1, w.d, w.tmp, NDT * DIM);
    k_rmsnorm<<<NDT, 256, 0, stream>>>(w.h, w.x1, dec[l].n2);
    run_mha(stream, dec[l].cattn, w.h, NDT, S_TGT, w.e, NE, S_SRC, src, 0, w.tmp, w);
    ew(NDT * DIM, g);
    k_add<<<g, 256, 0, stream>>>(w.x2, w.x1, w.tmp, NDT * DIM);
    k_rmsnorm<<<NDT, 256, 0, stream>>>(w.h, w.x2, dec[l].n3);
    if (dec[l].ff.is_moe) run_moe(stream, dec[l].ff.m, w.h, NDT, w.y, w);
    else                  run_ffn(stream, dec[l].ff.d, w.h, NDT, FFD, w.y, w);
    ew(NDT * DIM, g);
    k_add<<<g, 256, 0, stream>>>(w.d, w.x2, w.y, NDT * DIM);
  }

  // ---- final norm, output head, log-softmax ----
  k_rmsnorm<<<NDT, 256, 0, stream>>>(w.h, w.d, d_norm);
  gemm(stream, w.h, outp.w, outp.b, (float*)d_out, NDT, SMI_V, DIM);
  k_logsoftmax<<<NDT, 128, 0, stream>>>((float*)d_out);
}
